// NodeModel_31172872634968
// MI455X (gfx1250) — matrix operand order repaired and checked by reading, not compiled, not run on hardware
//
#include <hip/hip_runtime.h>
#include <hip/hip_bf16.h>

// ---------- types ----------
typedef __attribute__((ext_vector_type(16))) __bf16          v16bf;
typedef __attribute__((ext_vector_type(8)))  float           v8f;
typedef __attribute__((ext_vector_type(16))) unsigned short  v16u;
typedef __attribute__((ext_vector_type(8)))  unsigned short  v8u;

#define D_NODE   128
#define D_EDGE   128
#define D_GLOB   128
#define D_HID    256
#define D_OUT1   128
#define D_OUT2   128

// fp32 -> bf16 round-to-nearest-even (ignoring NaN corner)
__device__ __forceinline__ unsigned short f2bf(float f) {
    union { float f; unsigned u; } v; v.f = f;
    unsigned r = v.u + 0x7FFFu + ((v.u >> 16) & 1u);
    return (unsigned short)(r >> 16);
}

__device__ __forceinline__ ushort4 f4_to_bf4(float4 f) {
    ushort4 h;
    h.x = f2bf(f.x); h.y = f2bf(f.y); h.z = f2bf(f.z); h.w = f2bf(f.w);
    return h;
}

// A fragment (16-bit 16x32 layout): lane group 0 -> K {0..7,16..23}, group 1 -> K {8..15,24..31}
__device__ __forceinline__ v16bf loadA(const unsigned short* p) {
    union { v16u u; v8u h[2]; } t;
    t.h[0] = *(const v8u*)(p);
    t.h[1] = *(const v8u*)(p + 16);
    return *(v16bf*)&t.u;
}

// B fragment (16-bit 32x16 layout): lane group 0 -> K {0..7,16..23}, group 1 -> K {8..15,24..31}
__device__ __forceinline__ v16bf loadB(const unsigned short* p) {
    union { v16u u; v8u h[2]; } t;    // weights stored transposed [N][K]
    t.h[0] = *(const v8u*)(p);
    t.h[1] = *(const v8u*)(p + 16);
    return *(v16bf*)&t.u;
}

// ---------- weight convert: src [K][Nn] fp32 -> dst [Nn][K] bf16 ----------
__global__ __launch_bounds__(256) void convert_wt(const float* __restrict__ src,
                                                  unsigned short* __restrict__ dst,
                                                  int K, int Nn) {
    int i = blockIdx.x * 256 + threadIdx.x;
    if (i < K * Nn) {
        int k = i / Nn, n = i - k * Nn;
        dst[n * K + k] = f2bf(src[i]);
    }
}

__global__ __launch_bounds__(256) void zero_f32(float* __restrict__ p, int n) {
    int i = blockIdx.x * 256 + threadIdx.x;
    if (i < n) p[i] = 0.0f;
}

// ---------- CSR build (bandwidth-cheap replacement for 102M fp32 atomics) ----------
__global__ __launch_bounds__(256) void hist_dst(const int* __restrict__ ei,
                                                int* __restrict__ hist, int E) {
    int i = blockIdx.x * 256 + threadIdx.x;
    if (i < E) atomicAdd(&hist[ei[E + i]], 1);
}

// single-block chunked Hillis-Steele scan: hist[N] -> start[N+1], cursor[N]
__global__ __launch_bounds__(256) void scan_hist(const int* __restrict__ hist,
                                                 int* __restrict__ start,
                                                 int* __restrict__ cursor, int N) {
    __shared__ int buf[256];
    __shared__ int s_carry;
    int tid = threadIdx.x;
    if (tid == 0) s_carry = 0;
    __syncthreads();
    for (int base = 0; base < N; base += 256) {
        int v = (base + tid < N) ? hist[base + tid] : 0;
        buf[tid] = v;
        __syncthreads();
        for (int off = 1; off < 256; off <<= 1) {
            int t = (tid >= off) ? buf[tid - off] : 0;
            __syncthreads();
            buf[tid] += t;
            __syncthreads();
        }
        int excl  = buf[tid] - v;
        int carry = s_carry;
        if (base + tid < N) {
            start[base + tid]  = carry + excl;
            cursor[base + tid] = carry + excl;
        }
        int total = buf[255];
        __syncthreads();
        if (tid == 0) s_carry = carry + total;
        __syncthreads();
    }
    if (tid == 0) start[N] = s_carry;
}

__global__ __launch_bounds__(256) void scatter_eid(const int* __restrict__ ei,
                                                   int* __restrict__ cursor,
                                                   int* __restrict__ eidx, int E) {
    int i = blockIdx.x * 256 + threadIdx.x;
    if (i < E) {
        int pos = atomicAdd(&cursor[ei[E + i]], 1);
        eidx[pos] = i;
    }
}

// per-node coalesced gather-reduce: agg[n][c] = mean over incident edges of h[eid][c]
__global__ __launch_bounds__(128) void gather_agg(const float* __restrict__ h,
                                                  const int* __restrict__ start,
                                                  const int* __restrict__ eidx,
                                                  float* __restrict__ agg, int N) {
    int n = blockIdx.x;
    if (n >= N) return;
    int c = threadIdx.x;
    int s = start[n], e = start[n + 1];
    float acc = 0.0f;
    for (int j = s; j < e; ++j) acc += h[(size_t)eidx[j] * D_OUT1 + c];
    float d = (float)(e - s);
    agg[(size_t)n * D_OUT1 + c] = acc / (d > 1.0f ? d : 1.0f);
}

// ---------- edge MLP ----------
// block: 256 threads = 8 wave32, 64 edges. Wave owns N-tiles; B k-fragments cached in
// registers (weights read once per block). GEMM1 fused: one A pass feeds 2 accumulators.
// Output: hout != nullptr -> store h rows (CSR mode); else atomic scatter-add.
__global__ __launch_bounds__(256) void edge_mlp(
    const float* __restrict__ x, const int* __restrict__ ei,
    const float* __restrict__ ea,
    const unsigned short* __restrict__ w1a_t,   // [256][256] bf16
    const float* __restrict__ b1a,
    const unsigned short* __restrict__ w1b_t,   // [128][256] bf16
    const float* __restrict__ b1b,
    float* __restrict__ sum, float* __restrict__ cnt,
    float* __restrict__ hout, int E)
{
    const int SIN = 264, SH = 264;              // padded strides (bank rotation)
    __shared__ __align__(16) unsigned short s_in [64 * 264];
    __shared__ __align__(16) unsigned short s_hid[64 * 264];
    __shared__ int s_src[64], s_dst[64];

    const int tid = threadIdx.x;
    const int e0  = blockIdx.x * 64;

    if (tid < 64) {
        int e = e0 + tid, s = 0, d = 0;
        if (e < E) { s = ei[e]; d = ei[E + e]; }
        s_src[tid] = s; s_dst[tid] = d;
    }
    __syncthreads();

    // stage concat([x[row], edge_attr]) as bf16 [64 x 256], float4-vectorized
    for (int idx = tid; idx < 64 * 64; idx += 256) {
        int r = idx >> 6, c4 = (idx & 63) << 2;
        int e = e0 + r;
        float4 f = make_float4(0.f, 0.f, 0.f, 0.f);
        if (e < E) f = (c4 < D_NODE)
            ? *(const float4*)&x[s_src[r] * D_NODE + c4]
            : *(const float4*)&ea[e * D_EDGE + (c4 - D_NODE)];
        *(ushort4*)&s_in[r * SIN + c4] = f4_to_bf4(f);
    }
    __syncthreads();

    const int wid  = tid >> 5, lane = tid & 31;
    const int arow = lane & 15, akb = (lane >> 4) * 8;
    const int bcol = lane & 15, bks = (lane >> 4) * 8;
    const int mb   = (lane >> 4) * 8;

    // GEMM1: wave owns nt = wid and wid+8, fused over shared A fragments
    {
        const unsigned short* BbA = &w1a_t[(wid * 16 + bcol) * 256 + bks];
        const unsigned short* BbB = BbA + 128 * 256;        // nt + 8
        v16bf BfA[8], BfB[8];
        #pragma unroll
        for (int k8 = 0; k8 < 8; ++k8) { BfA[k8] = loadB(BbA + k8 * 32);
                                         BfB[k8] = loadB(BbB + k8 * 32); }
        float biasA = b1a[wid * 16 + bcol];
        float biasB = b1a[(wid + 8) * 16 + bcol];
        #pragma unroll
        for (int mt = 0; mt < 4; ++mt) {
            v8f a0 = {}, a1 = {};
            const unsigned short* Ab = &s_in[(mt * 16 + arow) * SIN + akb];
            #pragma unroll
            for (int k8 = 0; k8 < 8; ++k8) {
                v16bf A = loadA(Ab + k8 * 32);
                a0 = __builtin_amdgcn_wmma_f32_16x16x32_bf16(false, A, false, BfA[k8],
                                                             (short)0, a0, false, false);
                a1 = __builtin_amdgcn_wmma_f32_16x16x32_bf16(false, A, false, BfB[k8],
                                                             (short)0, a1, false, false);
            }
            #pragma unroll
            for (int v = 0; v < 8; ++v) {
                float r0 = a0[v] + biasA; r0 = r0 > 0.0f ? r0 : 0.0f;
                float r1 = a1[v] + biasB; r1 = r1 > 0.0f ? r1 : 0.0f;
                int rr = (mt * 16 + mb + v) * SH + bcol;
                s_hid[rr + wid * 16]       = f2bf(r0);
                s_hid[rr + (wid + 8) * 16] = f2bf(r1);
            }
        }
    }
    __syncthreads();

    // GEMM2: wave owns nt = wid; K=256
    {
        int nt = wid;
        const unsigned short* Bb = &w1b_t[(nt * 16 + bcol) * 256 + bks];
        v16bf Bf[8];
        #pragma unroll
        for (int k8 = 0; k8 < 8; ++k8) Bf[k8] = loadB(Bb + k8 * 32);
        float bias = b1b[nt * 16 + bcol];
        #pragma unroll
        for (int mt = 0; mt < 4; ++mt) {
            v8f acc = {};
            const unsigned short* Ab = &s_hid[(mt * 16 + arow) * SH + akb];
            #pragma unroll
            for (int k8 = 0; k8 < 8; ++k8) {
                v16bf A = loadA(Ab + k8 * 32);
                acc = __builtin_amdgcn_wmma_f32_16x16x32_bf16(false, A, false, Bf[k8],
                                                              (short)0, acc, false, false);
            }
            if (hout) {
                #pragma unroll
                for (int v = 0; v < 8; ++v) {
                    int lr = mt * 16 + mb + v;
                    int e  = e0 + lr;
                    if (e < E)
                        hout[(size_t)e * D_OUT1 + nt * 16 + bcol] = acc[v] + bias;
                }
            } else {
                #pragma unroll
                for (int v = 0; v < 8; ++v) {
                    int lr = mt * 16 + mb + v;
                    if (e0 + lr < E)
                        atomicAdd(&sum[s_dst[lr] * D_OUT1 + nt * 16 + bcol], acc[v] + bias);
                }
            }
        }
    }
    if (!hout && tid < 64 && e0 + tid < E) atomicAdd(&cnt[s_dst[tid]], 1.0f);
}

// ---------- node MLP ----------
// block: 256 threads = 8 wave32, 64 nodes. concat K=384 -> hid 256 -> out 128.
// cnt == nullptr means agg already contains the mean (CSR path).
__global__ __launch_bounds__(256) void node_mlp(
    const float* __restrict__ x, const float* __restrict__ u,
    const int* __restrict__ batch,
    const float* __restrict__ agg, const float* __restrict__ cnt,
    const unsigned short* __restrict__ w2a_t,   // [256][384] bf16
    const float* __restrict__ b2a,
    const unsigned short* __restrict__ w2b_t,   // [128][256] bf16
    const float* __restrict__ b2b,
    float* __restrict__ out, int N)
{
    const int SIN = 392, SH = 264;              // padded strides
    __shared__ __align__(16) unsigned short s_in [64 * 392];
    __shared__ __align__(16) unsigned short s_hid[64 * 264];
    __shared__ float s_inv[64];
    __shared__ int   s_b[64];

    const int tid = threadIdx.x;
    const int n0  = blockIdx.x * 64;

    if (tid < 64) {
        int n = n0 + tid; float inv = 0.0f; int b = 0;
        if (n < N) {
            if (cnt) {
                float c = cnt[n];
                inv = 1.0f / (c > 1.0f ? c : 1.0f);
            } else {
                inv = 1.0f;                      // agg already meaned
            }
            b = batch[n];
        }
        s_inv[tid] = inv; s_b[tid] = b;
    }
    __syncthreads();

    // stage concat([x, agg, u[batch]]) as bf16 [64 x 384], float4-vectorized
    for (int idx = tid; idx < 64 * 96; idx += 256) {
        int r = idx / 96, c4 = (idx - r * 96) << 2;
        int n = n0 + r;
        float4 f = make_float4(0.f, 0.f, 0.f, 0.f);
        if (n < N) {
            if (c4 < D_NODE) {
                f = *(const float4*)&x[n * D_NODE + c4];
            } else if (c4 < D_NODE + D_OUT1) {
                f = *(const float4*)&agg[(size_t)n * D_OUT1 + (c4 - D_NODE)];
                float s = s_inv[r];
                f.x *= s; f.y *= s; f.z *= s; f.w *= s;
            } else {
                f = *(const float4*)&u[s_b[r] * D_GLOB + (c4 - D_NODE - D_OUT1)];
            }
        }
        *(ushort4*)&s_in[r * SIN + c4] = f4_to_bf4(f);
    }
    __syncthreads();

    const int wid  = tid >> 5, lane = tid & 31;
    const int arow = lane & 15, akb = (lane >> 4) * 8;
    const int bcol = lane & 15, bks = (lane >> 4) * 8;
    const int mb   = (lane >> 4) * 8;

    // GEMM1: wave owns nt = wid and wid+8; K=384 (12 k-steps), B cached in registers
    #pragma unroll
    for (int j = 0; j < 2; ++j) {
        int nt = wid + j * 8;
        const unsigned short* Bb = &w2a_t[(nt * 16 + bcol) * 384 + bks];
        v16bf Bf[12];
        #pragma unroll
        for (int k8 = 0; k8 < 12; ++k8) Bf[k8] = loadB(Bb + k8 * 32);
        float bias = b2a[nt * 16 + bcol];
        #pragma unroll
        for (int mt = 0; mt < 4; ++mt) {
            v8f acc = {};
            const unsigned short* Ab = &s_in[(mt * 16 + arow) * SIN + akb];
            #pragma unroll
            for (int k8 = 0; k8 < 12; ++k8) {
                v16bf A = loadA(Ab + k8 * 32);
                acc = __builtin_amdgcn_wmma_f32_16x16x32_bf16(false, A, false, Bf[k8],
                                                              (short)0, acc, false, false);
            }
            #pragma unroll
            for (int v = 0; v < 8; ++v) {
                float r = acc[v] + bias;
                r = r > 0.0f ? r : 0.0f;
                s_hid[(mt * 16 + mb + v) * SH + nt * 16 + bcol] = f2bf(r);
            }
        }
    }
    __syncthreads();

    // GEMM2: wave owns nt = wid; K=256 -> final output (no ReLU)
    {
        int nt = wid;
        const unsigned short* Bb = &w2b_t[(nt * 16 + bcol) * 256 + bks];
        v16bf Bf[8];
        #pragma unroll
        for (int k8 = 0; k8 < 8; ++k8) Bf[k8] = loadB(Bb + k8 * 32);
        float bias = b2b[nt * 16 + bcol];
        #pragma unroll
        for (int mt = 0; mt < 4; ++mt) {
            v8f acc = {};
            const unsigned short* Ab = &s_hid[(mt * 16 + arow) * SH + akb];
            #pragma unroll
            for (int k8 = 0; k8 < 8; ++k8) {
                v16bf A = loadA(Ab + k8 * 32);
                acc = __builtin_amdgcn_wmma_f32_16x16x32_bf16(false, A, false, Bf[k8],
                                                              (short)0, acc, false, false);
            }
            #pragma unroll
            for (int v = 0; v < 8; ++v) {
                int n = n0 + mt * 16 + mb + v;
                if (n < N) out[n * D_OUT2 + nt * 16 + bcol] = acc[v] + bias;
            }
        }
    }
}

// ---------- launch ----------
extern "C" void kernel_launch(void* const* d_in, const int* in_sizes, int n_in,
                              void* d_out, int out_size, void* d_ws, size_t ws_size,
                              hipStream_t stream)
{
    const float* x     = (const float*)d_in[0];
    const int*   ei    = (const int*)  d_in[1];
    const float* ea    = (const float*)d_in[2];
    const float* u     = (const float*)d_in[3];
    const int*   batch = (const int*)  d_in[4];
    const float* W1a   = (const float*)d_in[5];
    const float* b1a   = (const float*)d_in[6];
    const float* W1b   = (const float*)d_in[7];
    const float* b1b   = (const float*)d_in[8];
    const float* W2a   = (const float*)d_in[9];
    const float* b2a   = (const float*)d_in[10];
    const float* W2b   = (const float*)d_in[11];
    const float* b2b   = (const float*)d_in[12];

    const int N = in_sizes[0] / D_NODE;      // 50000
    const int E = in_sizes[2] / D_EDGE;      // 800000

    // workspace layout (bytes)
    char* ws = (char*)d_ws;
    const size_t W_BYTES = 458752;                              // 4 bf16 weight mats
    unsigned short* w1a_t = (unsigned short*)(ws + 0);
    unsigned short* w1b_t = (unsigned short*)(ws + 131072);
    unsigned short* w2a_t = (unsigned short*)(ws + 196608);
    unsigned short* w2b_t = (unsigned short*)(ws + 393216);

    const size_t aggOff = W_BYTES;                              // N*128 fp32 (sum or mean)
    const size_t cntOff = aggOff + (size_t)N * D_OUT1 * 4;      // N fp32
    float* agg = (float*)(ws + aggOff);
    float* cnt = (float*)(ws + cntOff);

    // CSR extras
    const size_t histOff  = (cntOff + (size_t)N * 4 + 511) & ~(size_t)511;
    const size_t startOff = histOff  + (size_t)N * 4;
    const size_t curOff   = startOff + ((size_t)N + 1) * 4;
    const size_t eidxOff  = curOff   + (size_t)N * 4;
    const size_t hOff     = (eidxOff + (size_t)E * 4 + 511) & ~(size_t)511;
    const size_t csr_need = hOff + (size_t)E * D_OUT1 * 4;      // ~440 MB
    const bool use_csr = (ws_size >= csr_need);

    int* hist   = (int*)(ws + histOff);
    int* startp = (int*)(ws + startOff);
    int* cursor = (int*)(ws + curOff);
    int* eidx   = (int*)(ws + eidxOff);
    float* hbuf = (float*)(ws + hOff);

    // convert + transpose weights to bf16 [N][K]
    convert_wt<<<(256 * 256 + 255) / 256, 256, 0, stream>>>(W1a, w1a_t, 256, 256);
    convert_wt<<<(256 * 128 + 255) / 256, 256, 0, stream>>>(W1b, w1b_t, 256, 128);
    convert_wt<<<(384 * 256 + 255) / 256, 256, 0, stream>>>(W2a, w2a_t, 384, 256);
    convert_wt<<<(256 * 128 + 255) / 256, 256, 0, stream>>>(W2b, w2b_t, 256, 128);

    if (use_csr) {
        // CSR path: no fp32 atomics on the 128-wide payload
        zero_f32<<<(N + 255) / 256, 256, 0, stream>>>((float*)hist, N);
        hist_dst   <<<(E + 255) / 256, 256, 0, stream>>>(ei, hist, E);
        scan_hist  <<<1, 256, 0, stream>>>(hist, startp, cursor, N);
        scatter_eid<<<(E + 255) / 256, 256, 0, stream>>>(ei, cursor, eidx, E);
        edge_mlp   <<<(E + 63) / 64, 256, 0, stream>>>(
            x, ei, ea, w1a_t, b1a, w1b_t, b1b, nullptr, nullptr, hbuf, E);
        gather_agg <<<N, 128, 0, stream>>>(hbuf, startp, eidx, agg, N);
        node_mlp   <<<(N + 63) / 64, 256, 0, stream>>>(
            x, u, batch, agg, nullptr, w2a_t, b2a, w2b_t, b2b, (float*)d_out, N);
    } else {
        // fallback: atomic scatter-mean
        int zn = N * (D_OUT1 + 1);                  // agg ‖ cnt contiguous
        zero_f32<<<(zn + 255) / 256, 256, 0, stream>>>(agg, zn);
        edge_mlp<<<(E + 63) / 64, 256, 0, stream>>>(
            x, ei, ea, w1a_t, b1a, w1b_t, b1b, agg, cnt, nullptr, E);
        node_mlp<<<(N + 63) / 64, 256, 0, stream>>>(
            x, u, batch, agg, cnt, w2a_t, b2a, w2b_t, b2b, (float*)d_out, N);
    }
}
